// INT4Linear_62328565400263
// MI455X (gfx1250) — matrix operand order repaired and checked by reading, not compiled, not run on hardware
//
#include <hip/hip_runtime.h>
#include <stdint.h>

// INT4-weight linear: C[M,N] = A[M,K] * W[N,K]^T + bias
// M = 8192, K = 4096, N = 4096.
//   1) dequant INT4 -> fp16 into d_ws once (32 MiB, L2-resident)
//   2) f16 WMMA GEMM: 256x128x64 block tile, 8 wave32s, 64x64 per wave
//      (4x4 WMMA frags), HBM->LDS via global_load_async_to_lds_b128 with
//      double-buffered ASYNCcnt pipeline.

typedef __attribute__((ext_vector_type(16))) _Float16 v16h;
typedef __attribute__((ext_vector_type(8)))  _Float16 v8h;
typedef __attribute__((ext_vector_type(8)))  float    v8f;

static constexpr int Kdim = 4096;   // reduction dim
static constexpr int Odim = 4096;   // output features

// ---------------------------------------------------------------------------
// CDNA5 async HBM->LDS helpers
// ---------------------------------------------------------------------------
__device__ __forceinline__ void async_b128(uint32_t lds_off, const void* gptr) {
    asm volatile("global_load_async_to_lds_b128 %0, %1, off"
                 :: "v"(lds_off), "v"(gptr) : "memory");
}

template <int N>
__device__ __forceinline__ void wait_asynccnt() {
#if __has_builtin(__builtin_amdgcn_s_wait_asynccnt)
    __builtin_amdgcn_s_wait_asynccnt(N);
#else
    asm volatile("s_wait_asynccnt %0" :: "n"(N) : "memory");
#endif
}

// Flat pointer to __shared__: low 32 bits = LDS byte offset.
__device__ __forceinline__ uint32_t lds_offset(const void* p) {
    return (uint32_t)(uintptr_t)p;
}

// ---------------------------------------------------------------------------
// Kernel 1: INT4 -> fp16 weight dequant (memory-bound, ~40 MB traffic)
// ---------------------------------------------------------------------------
__global__ __launch_bounds__(256)
void dequant_int4_to_f16(const int8_t* __restrict__ wq,
                         const _Float16* __restrict__ wscale,
                         _Float16* __restrict__ w16)
{
    const int t = blockIdx.x * blockDim.x + threadIdx.x;   // one thread / 8 bytes
    const _Float16 sc = wscale[0];
    const size_t base = (size_t)t * 8;

    uint2 w = *(const uint2*)(wq + base);
    uint32_t words[2] = { w.x, w.y };
    v8h o0, o1;
    #pragma unroll
    for (int b = 0; b < 8; ++b) {
        uint32_t byte = (words[b >> 2] >> ((b & 3) * 8)) & 0xFFu;
        int hi = (int)((int8_t)byte) >> 4;           // signed high nibble -> even k
        int lo = (int)((byte & 0xFu) ^ 8u) - 8;      // signed low nibble  -> odd k
        _Float16 vh = (_Float16)hi * sc;
        _Float16 vl = (_Float16)lo * sc;
        if (b < 4) { o0[2 * b] = vh; o0[2 * b + 1] = vl; }
        else       { o1[2 * (b - 4)] = vh; o1[2 * (b - 4) + 1] = vl; }
    }
    *(v8h*)(w16 + base * 2)     = o0;
    *(v8h*)(w16 + base * 2 + 8) = o1;
}

// ---------------------------------------------------------------------------
// Kernel 2: f16 WMMA GEMM with async-to-LDS double buffering
// Block 256(M) x 128(N) x 64(K); wave grid 4x2; wave tile 64x64 (4x4 frags).
// ---------------------------------------------------------------------------
#define GM 256
#define GN 128
#define GK 64
#define NTHREADS 256

__global__ __launch_bounds__(NTHREADS)
void gemm_f16_wmma_async(const _Float16* __restrict__ x,
                         const _Float16* __restrict__ w16,
                         const _Float16* __restrict__ bias,
                         _Float16* __restrict__ out)
{
    __shared__ __align__(128) _Float16 Asm[2][GM][GK];   // 2 x 32 KB
    __shared__ __align__(128) _Float16 Bsm[2][GN][GK];   // 2 x 16 KB

    const int tid  = threadIdx.x;
    const int lane = tid & 31;
    const int wave = tid >> 5;
    const int waveM = wave >> 1;          // 0..3 -> 64-row strip
    const int waveN = wave & 1;           // 0..1 -> 64-col strip

    const int mTile = blockIdx.y * GM;
    const int nTile = blockIdx.x * GN;

    const uint32_t aLds0 = lds_offset(&Asm[0][0][0]);
    const uint32_t bLds0 = lds_offset(&Bsm[0][0][0]);

    // A tile: 2048 x 16B chunks; B tile: 1024 x 16B chunks.
    // chunk c: row = c>>3, k-offset = (c&7)*8 halfs; LDS offset = c*16 bytes
    // (row-major [row][k], 128 B/row -> global and LDS layouts coincide).
    auto stage = [&](int kt, int buf) {
        const _Float16* aBase = x   + (size_t)mTile * Kdim + kt * GK;
        const _Float16* bBase = w16 + (size_t)nTile * Kdim + kt * GK;
        const uint32_t aBuf = aLds0 + (uint32_t)buf * (GM * GK * 2);
        const uint32_t bBuf = bLds0 + (uint32_t)buf * (GN * GK * 2);
        #pragma unroll
        for (int i = 0; i < 8; ++i) {     // A: 8 chunks/thread
            const int c   = tid + 256 * i;
            const int row = c >> 3;
            const int k8  = (c & 7) * 8;
            async_b128(aBuf + (uint32_t)c * 16, aBase + (size_t)row * Kdim + k8);
        }
        #pragma unroll
        for (int i = 0; i < 4; ++i) {     // B: 4 chunks/thread
            const int c   = tid + 256 * i;
            const int row = c >> 3;
            const int k8  = (c & 7) * 8;
            async_b128(bBuf + (uint32_t)c * 16, bBase + (size_t)row * Kdim + k8);
        }
    };

    v8f acc[4][4] = {};                   // 128 VGPRs of f32 accumulators

    const int NK = Kdim / GK;             // 64 stages
    stage(0, 0);
    stage(1, 1);

    const int mr     = lane & 15;
    const int half16 = (lane >> 4) * 16;      // A f16 layout: K 0-15 / 16-31 per lane half
    const int kOffB  = (lane >> 4) * 16;      // B f16 layout: K 0-15 / 16-31 per lane half

    for (int kt = 0; kt < NK; ++kt) {
        const int cur = kt & 1;
        // stages kt and kt+1 in flight (12 asyncs each); leave next stage pending
        if (kt + 1 < NK) wait_asynccnt<12>();
        else             wait_asynccnt<0>();
        __syncthreads();                  // all waves' buf[cur] resident

        #pragma unroll
        for (int ks = 0; ks < 2; ++ks) {  // two WMMA k-steps per stage
            // B fragments stay resident for the whole k-step (4 x v16h)
            v16h bfr[4];
            #pragma unroll
            for (int j = 0; j < 4; ++j) {
                const int col = waveN * 64 + j * 16 + mr;
                bfr[j] = *(const v16h*)&Bsm[cur][col][ks * 32 + kOffB];
            }
            // stream A fragments, 4 WMMAs each
            #pragma unroll
            for (int i = 0; i < 4; ++i) {
                const int row = waveM * 64 + i * 16 + mr;
                v8h loh = *(const v8h*)&Asm[cur][row][ks * 32 + half16];
                v8h hih = *(const v8h*)&Asm[cur][row][ks * 32 + 8 + half16];
                v16h afr = __builtin_shufflevector(loh, hih,
                             0,1,2,3,4,5,6,7,8,9,10,11,12,13,14,15);
                #pragma unroll
                for (int j = 0; j < 4; ++j)
                    acc[i][j] = __builtin_amdgcn_wmma_f32_16x16x32_f16(
                        false, afr, false, bfr[j],
                        (short)0, acc[i][j], false, false);
            }
        }
        __syncthreads();                  // buf[cur] free to overwrite
        if (kt + 2 < NK) stage(kt + 2, cur);
    }

    // epilogue: C layout VGPR r -> M = r + 8*(lane/16), N = lane%16
    #pragma unroll
    for (int j = 0; j < 4; ++j) {
        const int col = nTile + waveN * 64 + j * 16 + (lane & 15);
        const float bv = (float)bias[col];
        #pragma unroll
        for (int i = 0; i < 4; ++i) {
            const int rowBase = mTile + waveM * 64 + i * 16 + ((lane >> 4) << 3);
            #pragma unroll
            for (int r = 0; r < 8; ++r)
                out[(size_t)(rowBase + r) * Odim + col] =
                    (_Float16)(acc[i][j][r] + bv);
        }
    }
}

// ---------------------------------------------------------------------------
// Fallback: fused dequant GEMM (128x128x32) if workspace is too small.
// ---------------------------------------------------------------------------
#define FM 128
#define FN 128
#define FK 32
__global__ __launch_bounds__(NTHREADS)
void int4_gemm_wmma_fused(const _Float16* __restrict__ x,
                          const int8_t*  __restrict__ wq,
                          const _Float16* __restrict__ wscale,
                          const _Float16* __restrict__ bias,
                          _Float16* __restrict__ out)
{
    __shared__ __align__(64) _Float16 Af[FM][FK];
    __shared__ __align__(64) _Float16 Bf[FN][FK];

    const int tid  = threadIdx.x;
    const int lane = tid & 31;
    const int wave = tid >> 5;
    const int waveM = wave >> 2;
    const int waveN = wave & 3;
    const int mTile = blockIdx.y * FM;
    const int nTile = blockIdx.x * FN;
    const _Float16 sc = wscale[0];

    const int aRow0 = tid >> 2;
    const int aCol0 = (tid & 3) * 8;
    const int wRow  = tid >> 1;
    const int wCol8 = (tid & 1) * 8;

    uint4 aReg0, aReg1;
    uint32_t wRegX, wRegY;
    auto loadTile = [&](int kt) {
        const _Float16* aBase = x + (size_t)mTile * Kdim + kt * FK;
        aReg0 = *(const uint4*)(aBase + (size_t)aRow0        * Kdim + aCol0);
        aReg1 = *(const uint4*)(aBase + (size_t)(aRow0 + 64) * Kdim + aCol0);
        const int8_t* wBase = wq + (size_t)(nTile + wRow) * (Kdim / 2)
                                 + kt * (FK / 2) + wCol8;
        uint2 w = *(const uint2*)wBase;
        wRegX = w.x; wRegY = w.y;
    };

    const int NK = Kdim / FK;
    loadTile(0);
    v8f acc[4][2] = {};

    for (int kt = 0; kt < NK; ++kt) {
        __syncthreads();
        *(uint4*)&Af[aRow0][aCol0]      = aReg0;
        *(uint4*)&Af[aRow0 + 64][aCol0] = aReg1;
        {
            uint32_t words[2] = { wRegX, wRegY };
            _Float16 v[16];
            #pragma unroll
            for (int b = 0; b < 8; ++b) {
                uint32_t byte = (words[b >> 2] >> ((b & 3) * 8)) & 0xFFu;
                int hi = (int)((int8_t)byte) >> 4;
                int lo = (int)((byte & 0xFu) ^ 8u) - 8;
                v[2 * b + 0] = (_Float16)hi * sc;
                v[2 * b + 1] = (_Float16)lo * sc;
            }
            v8h p0, p1;
            #pragma unroll
            for (int i = 0; i < 8; ++i) { p0[i] = v[i]; p1[i] = v[i + 8]; }
            *(v8h*)&Bf[wRow][wCol8 * 2 + 0] = p0;
            *(v8h*)&Bf[wRow][wCol8 * 2 + 8] = p1;
        }
        __syncthreads();
        if (kt + 1 < NK) loadTile(kt + 1);

        const int mr = lane & 15;
        const int half16 = (lane >> 4) * 16;
        v16h afr[4];
        #pragma unroll
        for (int i = 0; i < 4; ++i) {
            const int row = waveM * 64 + i * 16 + mr;
            v8h loh = *(const v8h*)&Af[row][half16];
            v8h hih = *(const v8h*)&Af[row][8 + half16];
            afr[i] = __builtin_shufflevector(loh, hih,
                       0,1,2,3,4,5,6,7,8,9,10,11,12,13,14,15);
        }
        v16h bfr[2];
        #pragma unroll
        for (int j = 0; j < 2; ++j) {
            const int col = waveN * 32 + j * 16 + mr;
            bfr[j] = *(const v16h*)&Bf[col][(lane >> 4) * 16];
        }
        #pragma unroll
        for (int i = 0; i < 4; ++i)
            #pragma unroll
            for (int j = 0; j < 2; ++j)
                acc[i][j] = __builtin_amdgcn_wmma_f32_16x16x32_f16(
                    false, afr[i], false, bfr[j], (short)0, acc[i][j], false, false);
    }

    #pragma unroll
    for (int j = 0; j < 2; ++j) {
        const int col = nTile + waveN * 32 + j * 16 + (lane & 15);
        const float bv = (float)bias[col];
        #pragma unroll
        for (int i = 0; i < 4; ++i) {
            const int rowBase = mTile + waveM * 64 + i * 16 + ((lane >> 4) << 3);
            #pragma unroll
            for (int r = 0; r < 8; ++r)
                out[(size_t)(rowBase + r) * Odim + col] =
                    (_Float16)(acc[i][j][r] + bv);
        }
    }
}

// ---------------------------------------------------------------------------
extern "C" void kernel_launch(void* const* d_in, const int* in_sizes, int n_in,
                              void* d_out, int out_size, void* d_ws, size_t ws_size,
                              hipStream_t stream) {
    const _Float16* x     = (const _Float16*)d_in[0];   // [B*S, K] fp16
    const int8_t*   wq    = (const int8_t*)  d_in[1];   // [O, K/2] packed int4
    const _Float16* scale = (const _Float16*)d_in[2];   // scalar fp16
    const _Float16* bias  = (const _Float16*)d_in[3];   // [O] fp16
    _Float16*       out   = (_Float16*)d_out;           // [B*S, O] fp16

    const int Mtotal = in_sizes[0] / Kdim;              // 8192

    const size_t needW16 = (size_t)Odim * Kdim * sizeof(_Float16);  // 32 MiB
    if (ws_size >= needW16 && (Mtotal % GM) == 0) {
        _Float16* w16 = (_Float16*)d_ws;
        const int nDeq = (Odim * (Kdim / 2)) / 8;       // threads, 8 bytes each
        dequant_int4_to_f16<<<nDeq / 256, 256, 0, stream>>>(wq, scale, w16);
        dim3 grid(Odim / GN, Mtotal / GM);              // (32, 32)
        gemm_f16_wmma_async<<<grid, NTHREADS, 0, stream>>>(x, w16, bias, out);
    } else {
        dim3 grid(Odim / FN, Mtotal / FM);              // (32, 64)
        int4_gemm_wmma_fused<<<grid, NTHREADS, 0, stream>>>(x, wq, scale, bias, out);
    }
}
